// TransformerBlock_7112465842427
// MI455X (gfx1250) — matrix operand order repaired and checked by reading, not compiled, not run on hardware
//
#include <hip/hip_runtime.h>

// ---------------------------------------------------------------------------
// Types for CDNA5 WMMA (wave32): V_WMMA_F32_16X16X32_BF16
// ---------------------------------------------------------------------------
typedef __attribute__((ext_vector_type(16))) __bf16 v16bf;
typedef __attribute__((ext_vector_type(8)))  __bf16 v8bf;
typedef __attribute__((ext_vector_type(8)))  float  v8f;

#define B_  8
#define T_  1024
#define C_  768
#define H_  12
#define D_  64
#define BT_ (B_ * T_)
#define NH_ 100
#define NHP_ 128   // padded hidden for WMMA

__device__ __forceinline__ v8f wmma_bf16(v16bf a, v16bf b, v8f c) {
  // 8 args: (neg_a, A, neg_b, B, c_mod, C, reuse_a, reuse_b)
  return __builtin_amdgcn_wmma_f32_16x16x32_bf16(false, a, false, b,
                                                 (short)0, c, false, false);
}

// A fragment: 16x32 bf16, row-major source, lane = M (mod 16),
// half = lane>>4 selects K sub-blocks {half*8..+7} and {16+half*8..+7}.
__device__ __forceinline__ v16bf load_a(const __bf16* A, int lda) {
  const int lane = threadIdx.x & 31;
  const __bf16* p = A + (size_t)(lane & 15) * lda + ((lane >> 4) << 3);
  v8bf lo = *(const v8bf*)(p);
  v8bf hi = *(const v8bf*)(p + 16);
  v16bf r;
#pragma unroll
  for (int i = 0; i < 8; ++i) { r[i] = lo[i]; r[8 + i] = hi[i]; }
  return r;
}

// B fragment: 32x16 bf16 from an N-major (transposed) source Bt[n][k]:
// lane = N (mod 16), half = lane>>4 selects K sub-blocks {half*8..+7} and {16+half*8..+7}.
__device__ __forceinline__ v16bf load_bt(const __bf16* Bt, int ldb) {
  const int lane = threadIdx.x & 31;
  const __bf16* p = Bt + (size_t)(lane & 15) * ldb + ((lane >> 4) << 3);
  v8bf lo = *(const v8bf*)(p);
  v8bf hi = *(const v8bf*)(p + 16);
  v16bf r;
#pragma unroll
  for (int i = 0; i < 8; ++i) { r[i] = lo[i]; r[8 + i] = hi[i]; }
  return r;
}

// Per-lane prefetch (lowers to global_prefetch_b8): lane covers one row.
__device__ __forceinline__ void pf_rows(const __bf16* base, int ld) {
  const int lane = threadIdx.x & 31;
  __builtin_prefetch(base + (size_t)lane * ld, 0, 3);
}

// ---------------------------------------------------------------------------
// Conversion kernels (one-time, tiny vs 402MB output traffic)
// ---------------------------------------------------------------------------
__global__ void cvt_f32_bf16(const float* __restrict__ in,
                             __bf16* __restrict__ out, int n) {
  int i = blockIdx.x * blockDim.x + threadIdx.x;
  if (i < n) out[i] = (__bf16)in[i];
}

// Bt[bi][n][kk] = (n<N && kk<K) ? W[bi][kk][n] : 0   (zero-pad to Kpad/Npad)
__global__ void cvt_bt(const float* __restrict__ W, __bf16* __restrict__ Bt,
                       int K, int N, int Kpad, int Npad, int batch) {
  long total = (long)batch * Npad * Kpad;
  long idx = (long)blockIdx.x * blockDim.x + threadIdx.x;
  if (idx >= total) return;
  int per = Npad * Kpad;
  int bi  = (int)(idx / per);
  int rem = (int)(idx % per);
  int n   = rem / Kpad;
  int kk  = rem % Kpad;
  float v = (n < N && kk < K) ? W[(size_t)bi * K * N + (size_t)kk * N + n] : 0.0f;
  Bt[idx] = (__bf16)v;
}

// ---------------------------------------------------------------------------
// QKV GEMM: X(8192x768) @ W[h](768x64) for 3 mats x 12 heads.
// grid = (64, 36); block = 256 (8 waves, 4x2 of 32x32 wave tiles).
// q scaled by 768^-0.5 (score scale folded in); v stored transposed (DxT).
// ---------------------------------------------------------------------------
__global__ __launch_bounds__(256)
void qkv_gemm(const __bf16* __restrict__ xb,
              const __bf16* __restrict__ Wqt, const __bf16* __restrict__ Wkt,
              const __bf16* __restrict__ Wvt,
              __bf16* __restrict__ qb, __bf16* __restrict__ kb,
              __bf16* __restrict__ vT) {
  const int zz = blockIdx.y;
  const int mat = zz / H_, h = zz % H_;
  const __bf16* Bt = (mat == 0 ? Wqt : (mat == 1 ? Wkt : Wvt)) + (size_t)h * D_ * C_;
  const int wave = threadIdx.x >> 5, lane = threadIdx.x & 31;
  const int m0 = blockIdx.x * 128 + (wave & 3) * 32;
  const int n0 = (wave >> 2) * 32;

  v8f acc[2][2] = {};
  for (int k0 = 0; k0 < C_; k0 += 32) {
    if (k0 + 32 < C_) pf_rows(xb + (size_t)m0 * C_ + k0 + 32, C_);
    v16bf a0 = load_a(xb + (size_t)m0 * C_ + k0, C_);
    v16bf a1 = load_a(xb + (size_t)(m0 + 16) * C_ + k0, C_);
    v16bf b0 = load_bt(Bt + (size_t)n0 * C_ + k0, C_);
    v16bf b1 = load_bt(Bt + (size_t)(n0 + 16) * C_ + k0, C_);
    acc[0][0] = wmma_bf16(a0, b0, acc[0][0]);
    acc[0][1] = wmma_bf16(a0, b1, acc[0][1]);
    acc[1][0] = wmma_bf16(a1, b0, acc[1][0]);
    acc[1][1] = wmma_bf16(a1, b1, acc[1][1]);
  }
  const int n = lane & 15, half = lane >> 4;
  const float qscale = 0.03608439182435161f;  // 768^-0.5
#pragma unroll
  for (int tm = 0; tm < 2; ++tm)
#pragma unroll
    for (int tn = 0; tn < 2; ++tn)
#pragma unroll
      for (int r = 0; r < 8; ++r) {
        int bt = m0 + tm * 16 + half * 8 + r;
        int d  = n0 + tn * 16 + n;
        int b  = bt >> 10, t = bt & (T_ - 1);
        size_t bh = (size_t)b * H_ + h;
        float v = acc[tm][tn][r];
        if (mat == 0)      qb[(bh * T_ + t) * D_ + d] = (__bf16)(v * qscale);
        else if (mat == 1) kb[(bh * T_ + t) * D_ + d] = (__bf16)v;
        else               vT[(bh * D_ + d) * T_ + t] = (__bf16)v;
      }
}

// ---------------------------------------------------------------------------
// Fused attention: per (b,h, 32-row q block): S = q kT (WMMA), softmax,
// write wei (f32 -> d_out, bf16 -> LDS), O = wei @ vT (WMMA) -> concat bf16.
// grid = (96, 32); block = 256; dyn LDS = 32*1024*4 + 32*1024*2 + 128 bytes.
// ---------------------------------------------------------------------------
__global__ __launch_bounds__(256)
void attn_kernel(const __bf16* __restrict__ qb, const __bf16* __restrict__ kb,
                 const __bf16* __restrict__ vT, float* __restrict__ wei_out,
                 __bf16* __restrict__ outb) {
  extern __shared__ char smem[];
  float*  S    = (float*)smem;                                // 32x1024 f32
  __bf16* Wb   = (__bf16*)(smem + 32 * 1024 * 4);             // 32x1024 bf16
  float*  reds = (float*)(smem + 32 * 1024 * 4 + 32 * 1024 * 2);  // 32 sums

  const int bh = blockIdx.x;            // b*H + h
  const int t0 = blockIdx.y * 32;
  const int b  = bh / H_, h = bh % H_;
  const int lane = threadIdx.x & 31, wave = threadIdx.x >> 5;

  const __bf16* qp = qb + ((size_t)bh * T_ + t0) * D_;
  const __bf16* kp = kb + (size_t)bh * T_ * D_;
  const __bf16* vp = vT + (size_t)bh * D_ * T_;

  // ---- phase 1: scores (2 x 64 tiles of 16x16; wave owns tm=w&1, tn=g+4j)
  {
    const int tm = wave & 1, g = wave >> 1;
    v16bf a0 = load_a(qp + tm * 16 * D_, D_);
    v16bf a1 = load_a(qp + tm * 16 * D_ + 32, D_);
    const int n = lane & 15, half = lane >> 4;
    for (int j = 0; j < 16; ++j) {
      int tn = g + 4 * j;
      if (j < 15) pf_rows(kp + (size_t)(tn + 4) * 16 * D_, D_);  // next k tile
      v16bf b0 = load_bt(kp + (size_t)tn * 16 * D_, D_);
      v16bf b1 = load_bt(kp + (size_t)tn * 16 * D_ + 32, D_);
      v8f acc = {};
      acc = wmma_bf16(a0, b0, acc);
      acc = wmma_bf16(a1, b1, acc);
#pragma unroll
      for (int r = 0; r < 8; ++r)
        S[(tm * 16 + half * 8 + r) * 1024 + tn * 16 + n] = acc[r];
    }
  }
  __syncthreads();

  // ---- phase 2: softmax (8 threads per row)
  {
    const int r = threadIdx.x >> 3, sub = threadIdx.x & 7;
    float* row = S + r * 1024;
    float m = -1e30f;
    for (int c = sub; c < 1024; c += 8) m = fmaxf(m, row[c]);
    m = fmaxf(m, __shfl_xor(m, 1));
    m = fmaxf(m, __shfl_xor(m, 2));
    m = fmaxf(m, __shfl_xor(m, 4));
    float ssum = 0.0f;
    for (int c = sub; c < 1024; c += 8) {
      float e = __expf(row[c] - m);
      row[c] = e;
      ssum += e;
    }
    ssum += __shfl_xor(ssum, 1);
    ssum += __shfl_xor(ssum, 2);
    ssum += __shfl_xor(ssum, 4);
    if (sub == 0) reds[r] = ssum;
  }
  __syncthreads();

  // normalize: coalesced f32 write of wei + bf16 copy for the V GEMM
  for (int idx = threadIdx.x; idx < 32 * 1024; idx += 256) {
    int r = idx >> 10, c = idx & 1023;
    float wv = S[idx] / reds[r];
    wei_out[((size_t)bh * T_ + t0 + r) * T_ + c] = wv;
    Wb[idx] = (__bf16)wv;
  }
  __syncthreads();

  // ---- phase 3: O = wei(32x1024) @ v(1024x64); 8 tiles, one per wave
  {
    const int tm = wave & 1, tn = wave >> 1;
    v8f acc = {};
    const __bf16* Abase = Wb + (size_t)tm * 16 * 1024;
    const __bf16* Bbase = vp + (size_t)tn * 16 * T_;
    for (int ks = 0; ks < 1024; ks += 32) {
      if (ks + 32 < 1024) pf_rows(Bbase + ks + 32, T_);  // next v tile
      v16bf a  = load_a(Abase + ks, 1024);
      v16bf bb = load_bt(Bbase + ks, T_);
      acc = wmma_bf16(a, bb, acc);
    }
    const int n = lane & 15, half = lane >> 4;
#pragma unroll
    for (int r = 0; r < 8; ++r) {
      int t = t0 + tm * 16 + half * 8 + r;
      int d = tn * 16 + n;
      outb[((size_t)b * T_ + t) * C_ + h * D_ + d] = (__bf16)acc[r];
    }
  }
}

// ---------------------------------------------------------------------------
// Generic bf16 GEMM, A(MxK) row-major, Bt(NxK) N-major; block tile 128x128,
// wave tile 32x64 (2x4 WMMA tiles => A-frag reused 4x, 8 indep. chains).
// MODE 0: Cf = acc + bias[col] (f32).  MODE 1: Cb = bf16(relu(acc+bias)).
// ---------------------------------------------------------------------------
template <int MODE>
__global__ __launch_bounds__(256)
void gemm_bt(const __bf16* __restrict__ A, int lda,
             const __bf16* __restrict__ Bt, int ldb,
             const float* __restrict__ bias, int biasN,
             float* __restrict__ Cf, __bf16* __restrict__ Cb, int ldc, int K) {
  const int wave = threadIdx.x >> 5, lane = threadIdx.x & 31;
  const int m0 = blockIdx.x * 128 + (wave & 3) * 32;
  const int n0 = blockIdx.y * 128 + (wave >> 2) * 64;

  v8f acc[2][4] = {};
  for (int k0 = 0; k0 < K; k0 += 32) {
    if (k0 + 32 < K) {
      pf_rows(A + (size_t)m0 * lda + k0 + 32, lda);         // next A tile (32 rows)
      pf_rows(Bt + (size_t)n0 * ldb + k0 + 32, ldb);        // next B rows n0..n0+31
      pf_rows(Bt + (size_t)(n0 + 32) * ldb + k0 + 32, ldb); // next B rows +32..+63
    }
    v16bf a0 = load_a(A + (size_t)m0 * lda + k0, lda);
    v16bf a1 = load_a(A + (size_t)(m0 + 16) * lda + k0, lda);
    v16bf bfrag[4];
#pragma unroll
    for (int j = 0; j < 4; ++j)
      bfrag[j] = load_bt(Bt + (size_t)(n0 + 16 * j) * ldb + k0, ldb);
#pragma unroll
    for (int j = 0; j < 4; ++j) {
      acc[0][j] = wmma_bf16(a0, bfrag[j], acc[0][j]);
      acc[1][j] = wmma_bf16(a1, bfrag[j], acc[1][j]);
    }
  }
  const int n = lane & 15, half = lane >> 4;
#pragma unroll
  for (int tm = 0; tm < 2; ++tm)
#pragma unroll
    for (int tn = 0; tn < 4; ++tn)
#pragma unroll
      for (int r = 0; r < 8; ++r) {
        int row = m0 + tm * 16 + half * 8 + r;
        int col = n0 + tn * 16 + n;
        float v = acc[tm][tn][r];
        if (MODE == 0) {
          if (bias) v += bias[col];
          Cf[(size_t)row * ldc + col] = v;
        } else {
          v += (col < biasN) ? bias[col] : 0.0f;
          v = v > 0.0f ? v : 0.0f;
          Cb[(size_t)row * ldc + col] = (__bf16)v;
        }
      }
}

// ---------------------------------------------------------------------------
// LayerNorm over C=768: out = LN(a + res) * g + beta. One 256-thread block/row.
// ---------------------------------------------------------------------------
__global__ __launch_bounds__(256)
void ln_kernel(const float* __restrict__ a, const float* __restrict__ res,
               const float* __restrict__ g, const float* __restrict__ beta,
               float* __restrict__ of, __bf16* __restrict__ ob) {
  __shared__ float red[256];
  const int row = blockIdx.x;
  const float* pa = a + (size_t)row * C_;
  const float* pr = res + (size_t)row * C_;
  float s[3];
  float sum = 0.0f;
#pragma unroll
  for (int i = 0; i < 3; ++i) {
    int c = threadIdx.x + i * 256;
    s[i] = pa[c] + pr[c];
    sum += s[i];
  }
  red[threadIdx.x] = sum;
  __syncthreads();
  for (int off = 128; off > 0; off >>= 1) {
    if (threadIdx.x < off) red[threadIdx.x] += red[threadIdx.x + off];
    __syncthreads();
  }
  float mu = red[0] * (1.0f / C_);
  __syncthreads();
  float vs = 0.0f;
#pragma unroll
  for (int i = 0; i < 3; ++i) { float d = s[i] - mu; vs += d * d; }
  red[threadIdx.x] = vs;
  __syncthreads();
  for (int off = 128; off > 0; off >>= 1) {
    if (threadIdx.x < off) red[threadIdx.x] += red[threadIdx.x + off];
    __syncthreads();
  }
  float inv = rsqrtf(red[0] * (1.0f / C_) + 1e-5f);
#pragma unroll
  for (int i = 0; i < 3; ++i) {
    int c = threadIdx.x + i * 256;
    float v = (s[i] - mu) * inv * g[c] + beta[c];
    if (of) of[(size_t)row * C_ + c] = v;
    if (ob) ob[(size_t)row * C_ + c] = (__bf16)v;
  }
}

// ---------------------------------------------------------------------------
extern "C" void kernel_launch(void* const* d_in, const int* in_sizes, int n_in,
                              void* d_out, int out_size, void* d_ws, size_t ws_size,
                              hipStream_t stream) {
  const float* x     = (const float*)d_in[0];
  const float* Wq    = (const float*)d_in[1];
  const float* Wk    = (const float*)d_in[2];
  const float* Wv    = (const float*)d_in[3];
  const float* Wproj = (const float*)d_in[4];
  const float* bproj = (const float*)d_in[5];
  const float* ln1_g = (const float*)d_in[6];
  const float* ln1_b = (const float*)d_in[7];
  const float* W1    = (const float*)d_in[8];
  const float* b1    = (const float*)d_in[9];
  const float* W2    = (const float*)d_in[10];
  const float* b2    = (const float*)d_in[11];
  const float* ln2_g = (const float*)d_in[12];
  const float* ln2_b = (const float*)d_in[13];

  float* wei = (float*)d_out;                               // (B,H,T,T)
  float* y   = wei + (size_t)B_ * H_ * T_ * T_;             // (B,T,C)

  // ---- workspace carve-up (all bf16/f32 scratch; ~133MB, reuses ff<-attn)
  char* ws = (char*)d_ws;
  size_t off = 0;
  auto alloc = [&](size_t bytes) -> char* {
    char* p = ws + off;
    off += (bytes + 255) & ~(size_t)255;
    return p;
  };
  __bf16* xb   = (__bf16*)alloc((size_t)BT_ * C_ * 2);
  __bf16* Wqt  = (__bf16*)alloc((size_t)H_ * D_ * C_ * 2);
  __bf16* Wkt  = (__bf16*)alloc((size_t)H_ * D_ * C_ * 2);
  __bf16* Wvt  = (__bf16*)alloc((size_t)H_ * D_ * C_ * 2);
  __bf16* Wpt  = (__bf16*)alloc((size_t)C_ * C_ * 2);
  __bf16* W1t  = (__bf16*)alloc((size_t)NHP_ * C_ * 2);
  __bf16* W2t  = (__bf16*)alloc((size_t)C_ * NHP_ * 2);
  __bf16* qb   = (__bf16*)alloc((size_t)B_ * H_ * T_ * D_ * 2);
  __bf16* kb   = (__bf16*)alloc((size_t)B_ * H_ * T_ * D_ * 2);
  __bf16* vT   = (__bf16*)alloc((size_t)B_ * H_ * D_ * T_ * 2);
  __bf16* outb = (__bf16*)alloc((size_t)BT_ * C_ * 2);
  float*  attn = (float*)alloc((size_t)BT_ * C_ * 4);       // also reused as ff
  float*  hf   = (float*)alloc((size_t)BT_ * C_ * 4);
  __bf16* hb   = (__bf16*)alloc((size_t)BT_ * C_ * 2);
  __bf16* hh   = (__bf16*)alloc((size_t)BT_ * NHP_ * 2);

  // 1) conversions / transposes
  {
    int n = BT_ * C_;
    cvt_f32_bf16<<<(n + 255) / 256, 256, 0, stream>>>(x, xb, n);
  }
  {
    long tot = (long)H_ * D_ * C_;
    cvt_bt<<<(unsigned)((tot + 255) / 256), 256, 0, stream>>>(Wq, Wqt, C_, D_, C_, D_, H_);
    cvt_bt<<<(unsigned)((tot + 255) / 256), 256, 0, stream>>>(Wk, Wkt, C_, D_, C_, D_, H_);
    cvt_bt<<<(unsigned)((tot + 255) / 256), 256, 0, stream>>>(Wv, Wvt, C_, D_, C_, D_, H_);
  }
  {
    long tot = (long)C_ * C_;
    cvt_bt<<<(unsigned)((tot + 255) / 256), 256, 0, stream>>>(Wproj, Wpt, C_, C_, C_, C_, 1);
  }
  {
    long tot = (long)NHP_ * C_;
    cvt_bt<<<(unsigned)((tot + 255) / 256), 256, 0, stream>>>(W1, W1t, C_, NH_, C_, NHP_, 1);
    cvt_bt<<<(unsigned)((tot + 255) / 256), 256, 0, stream>>>(W2, W2t, NH_, C_, NHP_, C_, 1);
  }

  // 2) QKV projections (WMMA)
  qkv_gemm<<<dim3(BT_ / 128, 3 * H_), 256, 0, stream>>>(xb, Wqt, Wkt, Wvt, qb, kb, vT);

  // 3) fused attention (WMMA scores + softmax + WMMA wei@V)
  {
    size_t smem = 32 * 1024 * 4 + 32 * 1024 * 2 + 128;  // 196736 B (< 320KB WGP LDS)
    attn_kernel<<<dim3(B_ * H_, T_ / 32), 256, smem, stream>>>(qb, kb, vT, wei, outb);
  }

  // 4) output projection + bias (block tile 128x128)
  gemm_bt<0><<<dim3(BT_ / 128, C_ / 128), 256, 0, stream>>>(
      outb, C_, Wpt, C_, bproj, C_, attn, nullptr, C_, C_);

  // 5) LN1(x + attn)
  ln_kernel<<<BT_, 256, 0, stream>>>(attn, x, ln1_g, ln1_b, hf, hb);

  // 6) FF1: relu(h @ W1 + b1) -> bf16 (padded N=128)
  gemm_bt<1><<<dim3(BT_ / 128, NHP_ / 128), 256, 0, stream>>>(
      hb, C_, W1t, C_, b1, NH_, nullptr, hh, NHP_, C_);

  // 7) FF2: hh @ W2 + b2 -> f32 (reuse attn buffer as ff)
  gemm_bt<0><<<dim3(BT_ / 128, C_ / 128), 256, 0, stream>>>(
      hh, NHP_, W2t, NHP_, b2, C_, attn, nullptr, C_, NHP_);

  // 8) LN2(h + ff) -> y
  ln_kernel<<<BT_, 256, 0, stream>>>(attn, hf, ln2_g, ln2_b, y, nullptr);
}
